// SoftKMeans_7876970021312
// MI455X (gfx1250) — matrix operand order repaired and checked by reading, not compiled, not run on hardware
//
#include <hip/hip_runtime.h>

// ---------------------------------------------------------------------------
// Soft K-Means, MI455X (gfx1250, wave32, WMMA).
// B=8, N=8192, D=256, K=256, ITERS=5, TEMP=0.5, EPS=1e-6.
// Compute-bound (86 GFLOP vs 64MB L2-resident data) -> bf16 WMMA GEMMs with
// fp32 accumulation; softmax statistics in fp32 VALU.  x2 cancels in softmax:
// logit_k = 4*xc - 2*c2_k  (since -dist2/T = -2*(x2 + c2 - 2xc)).
// ---------------------------------------------------------------------------

#define B_  8
#define N_  8192
#define D_  256
#define K_  256
#define ITERS_ 5
#define WT_PITCH 68   // bf16 elements; 68*2B = 34 dwords -> spreads LDS banks

typedef __attribute__((ext_vector_type(16))) __bf16 bf16x16;
typedef __attribute__((ext_vector_type(8)))  float  f32x8;

__device__ __forceinline__ f32x8 zero8() {
    f32x8 z;
#pragma unroll
    for (int i = 0; i < 8; ++i) z[i] = 0.0f;
    return z;
}

// A-operand (16-bit 16x32): lane l (row M=l&15, h=l>>4) holds K = {16h..16h+15}
// as 16 contiguous elements.
__device__ __forceinline__ bf16x16 load_a_from_f32(const float* __restrict__ p) {
    bf16x16 r;
#pragma unroll
    for (int i = 0; i < 8; ++i) r[i] = (__bf16)p[i];
#pragma unroll
    for (int i = 0; i < 8; ++i) r[8 + i] = (__bf16)p[8 + i];
    return r;
}

__device__ __forceinline__ bf16x16 load_a_from_bf16(const __bf16* p) {
    bf16x16 r;
#pragma unroll
    for (int i = 0; i < 8; ++i) r[i] = p[i];
#pragma unroll
    for (int i = 0; i < 8; ++i) r[8 + i] = p[8 + i];
    return r;
}

// B-operand (16-bit 32x16): lane l (col N=l&15, h=l>>4) holds inner rows
// K = {16h .. 16h+15} as 16 contiguous bf16.
__device__ __forceinline__ bf16x16 load_b_bf16(const __bf16* p) {
    bf16x16 r;
#pragma unroll
    for (int i = 0; i < 16; ++i) r[i] = p[i];
    return r;
}

__device__ __forceinline__ f32x8 wmma_bf16(bf16x16 a, bf16x16 b, f32x8 c) {
    return __builtin_amdgcn_wmma_f32_16x16x32_bf16(
        /*neg_a=*/false, a, /*neg_b=*/false, b,
        /*c_mod=*/(short)0, c, /*reuse_a=*/false, /*reuse_b=*/false);
}

// ---------------------------------------------------------------------------
// Kernel 0: x (B,N,D) f32 -> xT (B,D,N) bf16, 32x32 tiles via LDS.
// ---------------------------------------------------------------------------
__global__ __launch_bounds__(256) void xT_kernel(const float* __restrict__ x,
                                                 __bf16* __restrict__ xT) {
    const int bidx = blockIdx.x;                 // B * (N/32) * (D/32)
    const int tilesD = D_ / 32;                  // 8
    const int tilesN = N_ / 32;                  // 256
    const int b  = bidx / (tilesN * tilesD);
    const int r  = bidx % (tilesN * tilesD);
    const int tn = r / tilesD;
    const int td = r % tilesD;

    __shared__ __bf16 tile[32][33];
    const int col = threadIdx.x & 31;
    const int rr  = threadIdx.x >> 5;            // 0..7
#pragma unroll
    for (int i = 0; i < 4; ++i) {
        const int row = rr + i * 8;
        const float v = x[((size_t)b * N_ + tn * 32 + row) * D_ + td * 32 + col];
        tile[row][col] = (__bf16)v;
    }
    __syncthreads();
#pragma unroll
    for (int i = 0; i < 4; ++i) {
        const int drow = rr + i * 8;
        xT[((size_t)b * D_ + td * 32 + drow) * N_ + tn * 32 + col] = tile[col][drow];
    }
}

// ---------------------------------------------------------------------------
// Kernel 1: gather initial centroids -> c_bf (bf16) and c2 (f32).
// One block per (b,k); 256 threads = D.
// ---------------------------------------------------------------------------
__global__ __launch_bounds__(256) void init_kernel(const float* __restrict__ x,
                                                   const int* __restrict__ idx,
                                                   __bf16* __restrict__ c_bf,
                                                   float* __restrict__ c2g) {
    const int bk = blockIdx.x;
    const int b  = bk >> 8;
    const int k  = bk & 255;
    const int d  = threadIdx.x;
    const int n  = idx[b * K_ + k];
    const float v   = x[((size_t)b * N_ + n) * D_ + d];
    const __bf16 vb = (__bf16)v;
    c_bf[((size_t)b * K_ + k) * D_ + d] = vb;
    const float vr = (float)vb;

    __shared__ float red[256];
    red[d] = vr * vr;
    __syncthreads();
#pragma unroll
    for (int s = 128; s > 0; s >>= 1) {
        if (d < s) red[d] += red[d + s];
        __syncthreads();
    }
    if (d == 0) c2g[b * K_ + k] = red[0];
}

// ---------------------------------------------------------------------------
// Kernel 2 (fused, per iteration): one workgroup = (b, 64-row tile of N).
//   GEMM1 (WMMA bf16): XC[64,256] = Xtile[64,256] x Cbf^T
//   softmax over K (in-register + LDS cross-wave exchange)
//   GEMM2 (WMMA bf16): c_acc += W^T[256,64] x Xtile[64,256]  (f32 atomics)
// ---------------------------------------------------------------------------
__global__ __launch_bounds__(256) void
softkmeans_main(const float* __restrict__ x, const __bf16* __restrict__ xT,
                const __bf16* __restrict__ c_bf, const float* __restrict__ c2g,
                float* __restrict__ c_acc, float* __restrict__ denomg) {
    const int b    = blockIdx.x >> 7;            // N/64 = 128 tiles per b
    const int tile = blockIdx.x & 127;
    const int n0   = tile * 64;

    const int tid  = threadIdx.x;
    const int wave = tid >> 5;                   // 0..7 (wave32)
    const int lane = tid & 31;
    const int l15  = lane & 15;
    const int hx   = lane >> 4;

    const int strip = wave >> 1;                 // row strip 0..3 (16 rows each)
    const int kh    = wave & 1;                  // K half (8 k-tiles each)

    __shared__ float  sC2[K_];
    __shared__ float  sMax[64 * 2];
    __shared__ float  sSum[64 * 2];
    __shared__ float  sDen[4 * K_];
    __shared__ __bf16 sWT[K_ * WT_PITCH];        // W^T: [k][n], padded pitch

    sC2[tid] = c2g[b * K_ + tid];
    __syncthreads();

    // ---------------- GEMM1: logits = Xtile x Cbf^T ------------------------
    f32x8 acc[8];
#pragma unroll
    for (int kk = 0; kk < 8; ++kk) acc[kk] = zero8();

    const int rowg = n0 + strip * 16 + l15;      // global x row for A operand
#pragma unroll
    for (int dc = 0; dc < 8; ++dc) {             // D in chunks of 32
        const float* ap = x + ((size_t)b * N_ + rowg) * D_ + dc * 32 + 16 * hx;
        const bf16x16 a = load_a_from_f32(ap);
#pragma unroll
        for (int kk = 0; kk < 8; ++kk) {
            const int kt = kh * 8 + kk;          // 16-wide centroid tile
            const __bf16* bp = c_bf + ((size_t)b * K_ + kt * 16 + l15) * D_
                                    + dc * 32 + 16 * hx;
            acc[kk] = wmma_bf16(a, load_b_bf16(bp), acc[kk]);
        }
    }

    // ---------------- softmax over K (logit = 4*xc - 2*c2) -----------------
    float c2r[8];
#pragma unroll
    for (int kk = 0; kk < 8; ++kk) c2r[kk] = sC2[(kh * 8 + kk) * 16 + l15];
#pragma unroll
    for (int kk = 0; kk < 8; ++kk)
#pragma unroll
        for (int j = 0; j < 8; ++j)
            acc[kk][j] = 4.0f * acc[kk][j] - 2.0f * c2r[kk];

    // per-row partial max over this wave's 128 k's (butterfly over 16 lanes)
#pragma unroll
    for (int j = 0; j < 8; ++j) {
        float m = acc[0][j];
#pragma unroll
        for (int kk = 1; kk < 8; ++kk) m = fmaxf(m, acc[kk][j]);
#pragma unroll
        for (int s = 8; s >= 1; s >>= 1) m = fmaxf(m, __shfl_xor(m, s, 32));
        if (l15 == 0) sMax[(strip * 16 + j + 8 * hx) * 2 + kh] = m;
    }
    __syncthreads();

    float rm[8];
#pragma unroll
    for (int j = 0; j < 8; ++j) {
        const int row = strip * 16 + j + 8 * hx;
        rm[j] = fmaxf(sMax[row * 2], sMax[row * 2 + 1]);
    }
#pragma unroll
    for (int kk = 0; kk < 8; ++kk)
#pragma unroll
        for (int j = 0; j < 8; ++j)
            acc[kk][j] = __expf(acc[kk][j] - rm[j]);

#pragma unroll
    for (int j = 0; j < 8; ++j) {
        float s = 0.0f;
#pragma unroll
        for (int kk = 0; kk < 8; ++kk) s += acc[kk][j];
#pragma unroll
        for (int m = 8; m >= 1; m >>= 1) s += __shfl_xor(s, m, 32);
        if (l15 == 0) sSum[(strip * 16 + j + 8 * hx) * 2 + kh] = s;
    }
    __syncthreads();

    float inv_[8];
#pragma unroll
    for (int j = 0; j < 8; ++j) {
        const int row = strip * 16 + j + 8 * hx;
        inv_[j] = 1.0f / (sSum[row * 2] + sSum[row * 2 + 1]);
    }

    // normalize, write W^T (bf16) to LDS, accumulate per-k denom partials
#pragma unroll
    for (int kk = 0; kk < 8; ++kk) {
        const int k = (kh * 8 + kk) * 16 + l15;
        float dsum = 0.0f;
#pragma unroll
        for (int j = 0; j < 8; ++j) {
            const float w = acc[kk][j] * inv_[j];
            dsum += w;
            const int row = strip * 16 + j + 8 * hx;
            sWT[k * WT_PITCH + row] = (__bf16)w;
        }
        dsum += __shfl_xor(dsum, 16, 32);        // combine lane halves (rows)
        if (hx == 0) sDen[strip * K_ + k] = dsum;
    }
    __syncthreads();

    {   // reduce 4 strips, one global f32 atomic per k per workgroup
        const float dtot = sDen[tid] + sDen[K_ + tid] + sDen[2 * K_ + tid] +
                           sDen[3 * K_ + tid];
        unsafeAtomicAdd(&denomg[b * K_ + tid], dtot);
    }

    // ---------------- GEMM2: c_acc += W^T x Xtile --------------------------
#pragma unroll
    for (int ksl = 0; ksl < 2; ++ksl) {
        const int ks = wave * 2 + ksl;           // 16-row k strip of output
        f32x8 acc2[16];
#pragma unroll
        for (int dt = 0; dt < 16; ++dt) acc2[dt] = zero8();
#pragma unroll
        for (int nc = 0; nc < 2; ++nc) {         // inner n in chunks of 32
            const __bf16* wp = &sWT[(ks * 16 + l15) * WT_PITCH + nc * 32 + 16 * hx];
            const bf16x16 a2 = load_a_from_bf16(wp);
#pragma unroll
            for (int dt = 0; dt < 16; ++dt) {
                const __bf16* xp = xT + ((size_t)b * D_ + dt * 16 + l15) * N_
                                      + n0 + nc * 32 + 16 * hx;
                acc2[dt] = wmma_bf16(a2, load_b_bf16(xp), acc2[dt]);
            }
        }
#pragma unroll
        for (int dt = 0; dt < 16; ++dt)
#pragma unroll
            for (int j = 0; j < 8; ++j) {
                const int krow = ks * 16 + j + 8 * hx;
                const int dcol = dt * 16 + l15;
                unsafeAtomicAdd(&c_acc[((size_t)b * K_ + krow) * D_ + dcol],
                                acc2[dt][j]);
            }
    }
}

// ---------------------------------------------------------------------------
// Kernel 3: c = acc/(denom+eps); refresh c_bf, c2; last iter writes d_out.
// ---------------------------------------------------------------------------
__global__ __launch_bounds__(256) void
finalize_kernel(const float* __restrict__ c_acc, const float* __restrict__ denomg,
                __bf16* __restrict__ c_bf, float* __restrict__ c2g,
                float* __restrict__ out, int write_out) {
    const int bk = blockIdx.x;
    const int b  = bk >> 8;
    const int k  = bk & 255;
    const int d  = threadIdx.x;
    const float den = denomg[b * K_ + k] + 1e-6f;
    const size_t o  = ((size_t)b * K_ + k) * D_ + d;
    const float v   = c_acc[o] / den;
    const __bf16 vb = (__bf16)v;
    c_bf[o] = vb;
    const float vr = (float)vb;

    __shared__ float red[256];
    red[d] = vr * vr;
    __syncthreads();
#pragma unroll
    for (int s = 128; s > 0; s >>= 1) {
        if (d < s) red[d] += red[d + s];
        __syncthreads();
    }
    if (d == 0) c2g[b * K_ + k] = red[0];
    if (write_out) out[o] = v;
}

// ---------------------------------------------------------------------------
extern "C" void kernel_launch(void* const* d_in, const int* in_sizes, int n_in,
                              void* d_out, int out_size, void* d_ws, size_t ws_size,
                              hipStream_t stream) {
    const float* x  = (const float*)d_in[0];
    const int*  idx = (const int*)d_in[1];
    float* out = (float*)d_out;

    // workspace layout (~35 MB)
    char* ws = (char*)d_ws;
    __bf16* xT    = (__bf16*)(ws);                         // B*D*N*2 = 33,554,432
    __bf16* c_bf  = (__bf16*)(ws + 33554432);              // B*K*D*2 =  1,048,576
    float*  c2g   = (float*)(ws + 34603008);               // B*K*4   =      8,192
    float*  c_acc = (float*)(ws + 34611200);               // B*K*D*4 =  2,097,152
    float*  denomg= (float*)(ws + 36708352);               // B*K*4   =      8,192

    xT_kernel  <<<B_ * (N_ / 32) * (D_ / 32), 256, 0, stream>>>(x, xT);
    init_kernel<<<B_ * K_,                    256, 0, stream>>>(x, idx, c_bf, c2g);

    for (int it = 0; it < ITERS_; ++it) {
        hipMemsetAsync(c_acc, 0, (size_t)B_ * K_ * D_ * 4 + (size_t)B_ * K_ * 4,
                       stream);   // c_acc + denom (contiguous)
        softkmeans_main<<<B_ * (N_ / 64), 256, 0, stream>>>(x, xT, c_bf, c2g,
                                                            c_acc, denomg);
        finalize_kernel<<<B_ * K_, 256, 0, stream>>>(c_acc, denomg, c_bf, c2g,
                                                     out, it == ITERS_ - 1 ? 1 : 0);
    }
}
